// NeighborEmbedder_35390530519811
// MI455X (gfx1250) — matrix operand order repaired and checked by reading, not compiled, not run on hardware
//
#include <hip/hip_runtime.h>

typedef __attribute__((ext_vector_type(16))) __bf16 v16bf;
typedef __attribute__((ext_vector_type(8)))  float  v8f;

// ---------- helpers ----------
__device__ __forceinline__ unsigned short f2bf_rhu(float x) {   // round-half-up
  return (unsigned short)((__float_as_uint(x) + 0x8000u) >> 16);
}
// pack two floats -> two bf16 in one dword via v_perm_b32 (lo = a, hi = b)
__device__ __forceinline__ unsigned pkbf(float a, float b) {
  unsigned ua = __float_as_uint(a) + 0x8000u;
  unsigned ub = __float_as_uint(b) + 0x8000u;
  return __builtin_amdgcn_perm(ub, ua, 0x07060302u);
}

union Frag { v16bf v; uint4 q[2]; unsigned u32[8]; };

// =====================================================================
// GEMM:  C[M x N] = A[M x K] @ W[N x K]^T (+bias +pos +resid, opt gelu)
// A given as bf16 (Ab) OR fp32 (Af); W pre-converted bf16.
// Outputs: optional fp32 (outF) and/or bf16 (outB), both M x N row-major.
// All call sites: M%128==0, N%128==0, K%32==0.
// Block = 256 thr (8 waves, 4x2), block tile 128x128, wave tile 32x64.
// =====================================================================
__global__ __launch_bounds__(256)
void gemm_bf16_wmma(const unsigned short* __restrict__ Ab,
                    const float* __restrict__ Af,
                    long long aStride, long long aOff,
                    const unsigned short* __restrict__ Wb,
                    const float* __restrict__ bias,
                    const float* __restrict__ resid, long long rStride, long long rOff,
                    const float* __restrict__ pos,      // (9,N): add pos[(row%9)*N+col]
                    float* __restrict__ outF,
                    unsigned short* __restrict__ outB,
                    int M, int N, int K, int fuseGelu) {
  const int lane = threadIdx.x & 31;
  const int wid  = threadIdx.x >> 5;       // 0..7
  const int m0 = blockIdx.y * 128 + (wid >> 1) * 32;
  const int n0 = blockIdx.x * 128 + (wid & 1) * 64;
  const int r  = lane & 15;
  const int h  = lane >> 4;                // 0 or 1

  v8f acc[2][4];
#pragma unroll
  for (int i = 0; i < 2; ++i)
#pragma unroll
    for (int j = 0; j < 4; ++j) acc[i][j] = (v8f)0.f;

  if (Ab) {
    // ---- fast path: A already bf16 -> pure b128 loads, no VALU packing ----
    for (int k0 = 0; k0 < K; k0 += 32) {
      Frag afrag[2];
#pragma unroll
      for (int mt = 0; mt < 2; ++mt) {
        const unsigned short* ap = Ab + aOff + (long long)(m0 + mt*16 + r) * aStride + k0;
        afrag[mt].q[0] = *(const uint4*)(ap + h*8);        // slots 0..7 : K=h*8..
        afrag[mt].q[1] = *(const uint4*)(ap + 16 + h*8);   // slots 8..15: K=16+h*8..
      }
      Frag bfrag[4];
#pragma unroll
      for (int nt = 0; nt < 4; ++nt) {
        const uint4* bp = (const uint4*)(Wb + (long long)(n0 + nt*16 + r) * K + k0 + h*8);
        bfrag[nt].q[0] = bp[0];                            // slots 0..7: K=h*8.., slots 8..15: K=16+h*8..
        bfrag[nt].q[1] = bp[2];
      }
#pragma unroll
      for (int mt = 0; mt < 2; ++mt)
#pragma unroll
        for (int nt = 0; nt < 4; ++nt)
          acc[mt][nt] = __builtin_amdgcn_wmma_f32_16x16x32_bf16(
              false, afrag[mt].v, false, bfrag[nt].v, (short)0, acc[mt][nt], false, false);
    }
  } else {
    // ---- A fp32 path (patch-embed only): convert with v_perm packing ----
    for (int k0 = 0; k0 < K; k0 += 32) {
      Frag afrag[2];
#pragma unroll
      for (int mt = 0; mt < 2; ++mt) {
        const float* ap = Af + aOff + (long long)(m0 + mt*16 + r) * aStride + k0 + h*8;
        float4 x0 = *(const float4*)(ap);
        float4 x1 = *(const float4*)(ap + 4);
        float4 x2 = *(const float4*)(ap + 16);
        float4 x3 = *(const float4*)(ap + 20);
        afrag[mt].u32[0] = pkbf(x0.x, x0.y); afrag[mt].u32[1] = pkbf(x0.z, x0.w);
        afrag[mt].u32[2] = pkbf(x1.x, x1.y); afrag[mt].u32[3] = pkbf(x1.z, x1.w);
        afrag[mt].u32[4] = pkbf(x2.x, x2.y); afrag[mt].u32[5] = pkbf(x2.z, x2.w);
        afrag[mt].u32[6] = pkbf(x3.x, x3.y); afrag[mt].u32[7] = pkbf(x3.z, x3.w);
      }
      Frag bfrag[4];
#pragma unroll
      for (int nt = 0; nt < 4; ++nt) {
        const uint4* bp = (const uint4*)(Wb + (long long)(n0 + nt*16 + r) * K + k0 + h*8);
        bfrag[nt].q[0] = bp[0];
        bfrag[nt].q[1] = bp[2];
      }
#pragma unroll
      for (int mt = 0; mt < 2; ++mt)
#pragma unroll
        for (int nt = 0; nt < 4; ++nt)
          acc[mt][nt] = __builtin_amdgcn_wmma_f32_16x16x32_bf16(
              false, afrag[mt].v, false, bfrag[nt].v, (short)0, acc[mt][nt], false, false);
    }
  }

  // epilogue: lane holds col = n0+nt*16+r; rows m0+mt*16 + g + 8*h
#pragma unroll
  for (int mt = 0; mt < 2; ++mt) {
#pragma unroll
    for (int nt = 0; nt < 4; ++nt) {
      const int col = n0 + nt*16 + r;
      const float bv = bias ? bias[col] : 0.f;
#pragma unroll
      for (int g = 0; g < 8; ++g) {
        const int row = m0 + mt*16 + h*8 + g;
        float v = acc[mt][nt][g] + bv;
        if (pos)   v += pos[(row % 9) * N + col];
        if (resid) v += resid[rOff + (long long)row * rStride + col];
        if (fuseGelu) v = 0.5f * v * (1.f + erff(v * 0.70710678118654752f));
        const long long idx = (long long)row * N + col;
        if (outF) outF[idx] = v;
        if (outB) outB[idx] = f2bf_rhu(v);
      }
    }
  }
}

// =====================================================================
// Elementwise fp32 -> bf16 (2 elements per thread, n even)
// =====================================================================
__global__ __launch_bounds__(256)
void f32_to_bf16_kernel(const float* __restrict__ in, unsigned* __restrict__ out,
                        long long npairs) {
  long long i = (long long)blockIdx.x * blockDim.x + threadIdx.x;
  if (i < npairs) out[i] = pkbf(in[2*i], in[2*i + 1]);
}

// =====================================================================
// LayerNorm over rows of width 512, one wave per row; fp32 + opt bf16 out.
// =====================================================================
__global__ __launch_bounds__(256)
void ln512_kernel(const float* __restrict__ in, const float* __restrict__ gamma,
                  const float* __restrict__ beta, float* __restrict__ out,
                  unsigned short* __restrict__ outB, int M) {
  const int lane = threadIdx.x & 31;
  const int row  = blockIdx.x * (blockDim.x >> 5) + (threadIdx.x >> 5);
  if (row >= M) return;
  const float* p = in + (size_t)row * 512;
  float v[16];
  float s = 0.f;
#pragma unroll
  for (int i = 0; i < 16; ++i) { v[i] = p[lane + 32*i]; s += v[i]; }
#pragma unroll
  for (int off = 16; off; off >>= 1) s += __shfl_xor(s, off, 32);
  const float mean = s * (1.f / 512.f);
  float q = 0.f;
#pragma unroll
  for (int i = 0; i < 16; ++i) { float d = v[i] - mean; q += d * d; }
#pragma unroll
  for (int off = 16; off; off >>= 1) q += __shfl_xor(q, off, 32);
  const float inv = rsqrtf(q * (1.f / 512.f) + 1e-5f);
  float* o = out + (size_t)row * 512;
  unsigned short* ob = outB ? outB + (size_t)row * 512 : nullptr;
#pragma unroll
  for (int i = 0; i < 16; ++i) {
    const int c = lane + 32*i;
    const float y = (v[i] - mean) * inv * gamma[c] + beta[c];
    o[c] = y;
    if (ob) ob[c] = f2bf_rhu(y);
  }
}

// =====================================================================
// Self-attention over 9 tokens, d=64. One wave per (seq, head).
// QKV rows 1536 wide: [q|k|v]. Output written as bf16 (next GEMM's A).
// =====================================================================
__global__ __launch_bounds__(32)
void sa_attn_kernel(const float* __restrict__ QKV, unsigned short* __restrict__ O) {
  __shared__ float sq[9][64], sk[9][64], sv[9][64], sc[9][9];
  const int lane = threadIdx.x;
  const int s  = blockIdx.x >> 3;
  const int hh = blockIdx.x & 7;
  const size_t base = (size_t)s * 9 * 1536 + hh * 64;
  for (int idx = lane; idx < 576; idx += 32) {
    const int t = idx >> 6, d = idx & 63;
    const size_t rb = base + (size_t)t * 1536 + d;
    sq[t][d] = QKV[rb];
    sk[t][d] = QKV[rb + 512];
    sv[t][d] = QKV[rb + 1024];
  }
  __syncthreads();
  for (int p = lane; p < 81; p += 32) {
    const int qi = p / 9, kj = p - qi * 9;
    float a = 0.f;
#pragma unroll
    for (int d = 0; d < 64; ++d) a += sq[qi][d] * sk[kj][d];
    sc[qi][kj] = a * 0.125f;     // 1/sqrt(64)
  }
  __syncthreads();
  if (lane < 9) {
    float m = sc[lane][0];
#pragma unroll
    for (int j = 1; j < 9; ++j) m = fmaxf(m, sc[lane][j]);
    float e[9], sum = 0.f;
#pragma unroll
    for (int j = 0; j < 9; ++j) { e[j] = __expf(sc[lane][j] - m); sum += e[j]; }
    const float inv = 1.f / sum;
#pragma unroll
    for (int j = 0; j < 9; ++j) sc[lane][j] = e[j] * inv;
  }
  __syncthreads();
  const size_t obase = (size_t)s * 9 * 512 + hh * 64;
#pragma unroll
  for (int qi = 0; qi < 9; ++qi) {
#pragma unroll
    for (int dd = 0; dd < 2; ++dd) {
      const int d = lane + dd * 32;
      float a = 0.f;
#pragma unroll
      for (int j = 0; j < 9; ++j) a += sc[qi][j] * sv[j][d];
      O[obase + (size_t)qi * 512 + d] = f2bf_rhu(a);
    }
  }
}

// =====================================================================
// Cross-attention: 1 query over 9 keys, d=64. One wave per (seq, head).
// KV rows 1024 wide: [k|v]. Output bf16.
// =====================================================================
__global__ __launch_bounds__(32)
void ca_attn_kernel(const float* __restrict__ Q, const float* __restrict__ KV,
                    unsigned short* __restrict__ O) {
  __shared__ float sk[9][64], sv[9][64], sq[64], sc[9];
  const int lane = threadIdx.x;
  const int s  = blockIdx.x >> 3;
  const int hh = blockIdx.x & 7;
  for (int idx = lane; idx < 64; idx += 32) sq[idx] = Q[(size_t)s * 512 + hh * 64 + idx];
  const size_t kb = (size_t)s * 9 * 1024 + hh * 64;
  for (int idx = lane; idx < 576; idx += 32) {
    const int t = idx >> 6, d = idx & 63;
    sk[t][d] = KV[kb + (size_t)t * 1024 + d];
    sv[t][d] = KV[kb + (size_t)t * 1024 + 512 + d];
  }
  __syncthreads();
  if (lane < 9) {
    float a = 0.f;
#pragma unroll
    for (int d = 0; d < 64; ++d) a += sq[d] * sk[lane][d];
    sc[lane] = a * 0.125f;
  }
  __syncthreads();
  float m = sc[0];
#pragma unroll
  for (int j = 1; j < 9; ++j) m = fmaxf(m, sc[j]);
  float e[9], sum = 0.f;
#pragma unroll
  for (int j = 0; j < 9; ++j) { e[j] = __expf(sc[j] - m); sum += e[j]; }
  const float inv = 1.f / sum;
#pragma unroll
  for (int dd = 0; dd < 2; ++dd) {
    const int d = lane + dd * 32;
    float a = 0.f;
#pragma unroll
    for (int j = 0; j < 9; ++j) a += e[j] * sv[j][d];
    O[(size_t)s * 512 + hh * 64 + d] = f2bf_rhu(a * inv);
  }
}

// =====================================================================
// Host-side orchestration
// =====================================================================
extern "C" void kernel_launch(void* const* d_in, const int* in_sizes, int n_in,
                              void* d_out, int out_size, void* d_ws, size_t ws_size,
                              hipStream_t stream) {
  (void)in_sizes; (void)n_in; (void)out_size; (void)ws_size;

  const float* P        = (const float*)d_in[0];   // (73728, 768)
  const float* w_pe     = (const float*)d_in[1];   // (512, 768)
  const float* b_pe     = (const float*)d_in[2];
  const float* pos      = (const float*)d_in[3];   // (9, 512)
  const float* sa_in_w  = (const float*)d_in[4];   // (1536, 512)
  const float* sa_in_b  = (const float*)d_in[5];
  const float* sa_out_w = (const float*)d_in[6];   // (512, 512)
  const float* sa_out_b = (const float*)d_in[7];
  const float* g1       = (const float*)d_in[8];
  const float* beta1    = (const float*)d_in[9];
  const float* ca_in_w  = (const float*)d_in[10];  // (1536, 512)
  const float* ca_in_b  = (const float*)d_in[11];
  const float* ca_out_w = (const float*)d_in[12];
  const float* ca_out_b = (const float*)d_in[13];
  const float* g2       = (const float*)d_in[14];
  const float* beta2    = (const float*)d_in[15];
  const float* w1       = (const float*)d_in[16];  // (2048, 512)
  const float* b1       = (const float*)d_in[17];
  const float* w2       = (const float*)d_in[18];  // (512, 2048)
  const float* b2       = (const float*)d_in[19];
  const float* g3       = (const float*)d_in[20];
  const float* beta3    = (const float*)d_in[21];

  // ---- workspace layout (bytes) ----
  char* wsb = (char*)d_ws;
  float*          Xf  = (float*)(wsb);                                   // 73728*512 f32  = 150,994,944 B
  unsigned short* Xb  = (unsigned short*)(wsb + 150994944LL);            // 73728*512 bf16 =  75,497,472 B
  float*          BUF = (float*)(wsb + 226492416LL);                     // 73728*1536 f32 = 452,984,832 B
  unsigned short* CBb = (unsigned short*)(wsb + 679477248LL);            // 73728*512 bf16 =  75,497,472 B
  unsigned short* WB  = (unsigned short*)(wsb + 754974720LL);            // 4,587,520 bf16 ~ 9.2 MB

  // bf16 weight sub-buffers (element offsets)
  unsigned short* wpe_b   = WB;                    // 512*768  = 393216
  unsigned short* sain_b  = WB + 393216;           // 1536*512 = 786432
  unsigned short* saout_b = WB + 1179648;          // 512*512  = 262144
  unsigned short* cain_b  = WB + 1441792;          // 1536*512 = 786432
  unsigned short* caout_b = WB + 2228224;          // 512*512  = 262144
  unsigned short* w1_b    = WB + 2490368;          // 2048*512 = 1048576
  unsigned short* w2_b    = WB + 3538944;          // 512*2048 = 1048576

  const dim3 blk(256);
  const long long S512 = 512, S9x512 = 9 * 512, OFF4 = 4 * 512;
  auto cvt = [&](const float* src, unsigned short* dst, long long n) {
    long long pairs = n / 2;
    f32_to_bf16_kernel<<<(unsigned)((pairs + 255) / 256), 256, 0, stream>>>(src, (unsigned*)dst, pairs);
  };

  // 0) pre-convert all weights to bf16 (read once per GEMM thereafter)
  cvt(w_pe,     wpe_b,   393216);
  cvt(sa_in_w,  sain_b,  786432);
  cvt(sa_out_w, saout_b, 262144);
  cvt(ca_in_w,  cain_b,  786432);
  cvt(ca_out_w, caout_b, 262144);
  cvt(w1,       w1_b,    1048576);
  cvt(w2,       w2_b,    1048576);

  // 1) X = P @ w_pe^T + b_pe + pos   (fp32-A path; dual write f32 + bf16)
  gemm_bf16_wmma<<<dim3(4, 576), blk, 0, stream>>>(
      nullptr, P, 768, 0, wpe_b, b_pe, nullptr, 0, 0, pos, Xf, Xb, 73728, 512, 768, 0);

  // 2) QKV = X @ sa_in_w^T + b        (73728 x 1536)
  gemm_bf16_wmma<<<dim3(12, 576), blk, 0, stream>>>(
      Xb, nullptr, S512, 0, sain_b, sa_in_b, nullptr, 0, 0, nullptr, BUF, nullptr,
      73728, 1536, 512, 0);

  // 3) self-attention -> O (bf16)
  sa_attn_kernel<<<65536, 32, 0, stream>>>(BUF, CBb);

  // 4) pre = X + O @ sa_out_w^T + b   (f32, into BUF)
  gemm_bf16_wmma<<<dim3(4, 576), blk, 0, stream>>>(
      CBb, nullptr, S512, 0, saout_b, sa_out_b, Xf, S512, 0, nullptr, BUF, nullptr,
      73728, 512, 512, 0);

  // 5) X1 = LN(pre)  -> f32 over Xf, bf16 over Xb
  ln512_kernel<<<9216, 256, 0, stream>>>(BUF, g1, beta1, Xf, Xb, 73728);

  // 6) KV = X1 @ [wk;wv]^T + b        (73728 x 1024, f32 in BUF)
  gemm_bf16_wmma<<<dim3(8, 576), blk, 0, stream>>>(
      Xb, nullptr, S512, 0, cain_b + 512LL * 512, ca_in_b + 512, nullptr, 0, 0, nullptr,
      BUF, nullptr, 73728, 1024, 512, 0);

  // 7) Qc = cq @ wq^T + bq  (cq = X1 rows 9s+4)  (8192 x 512)
  float* Qc = BUF + 75497472LL;
  gemm_bf16_wmma<<<dim3(4, 64), blk, 0, stream>>>(
      Xb, nullptr, S9x512, OFF4, cain_b, ca_in_b, nullptr, 0, 0, nullptr, Qc, nullptr,
      8192, 512, 512, 0);

  // 8) cross-attention -> O2 (bf16, at CBb offset 0)
  ca_attn_kernel<<<65536, 32, 0, stream>>>(Qc, BUF, CBb);

  // 9) pre2 = cq + O2 @ ca_out_w^T + b  (f32)
  float* pre2 = BUF;                       // KV consumed
  gemm_bf16_wmma<<<dim3(4, 64), blk, 0, stream>>>(
      CBb, nullptr, S512, 0, caout_b, ca_out_b, Xf, S9x512, OFF4, nullptr, pre2, nullptr,
      8192, 512, 512, 0);

  // 10) c = LN(pre2) -> f32 + bf16
  float*          c_f = BUF + 4194304LL;
  unsigned short* c_b = CBb + 4194304LL;
  ln512_kernel<<<1024, 256, 0, stream>>>(pre2, g2, beta2, c_f, c_b, 8192);

  // 11) h = gelu(c @ w1^T + b1)   (8192 x 2048, bf16-only output)
  unsigned short* h_b = CBb + 8388608LL;
  gemm_bf16_wmma<<<dim3(16, 64), blk, 0, stream>>>(
      c_b, nullptr, S512, 0, w1_b, b1, nullptr, 0, 0, nullptr, nullptr, h_b,
      8192, 2048, 512, 1);

  // 12) pre3 = c + h @ w2^T + b2  (8192 x 512, K=2048)
  float* pre3 = BUF + 8388608LL;
  gemm_bf16_wmma<<<dim3(4, 64), blk, 0, stream>>>(
      h_b, nullptr, 2048, 0, w2_b, b2, c_f, S512, 0, nullptr, pre3, nullptr,
      8192, 512, 2048, 0);

  // 13) out = LN(pre3)
  ln512_kernel<<<1024, 256, 0, stream>>>(pre3, g3, beta3, (float*)d_out, nullptr, 8192);
}
